// HybridOmniRobotSim_29729763623473
// MI455X (gfx1250) — matrix operand order repaired and checked by reading, not compiled, not run on hardware
//
#include <hip/hip_runtime.h>
#include <hip/hip_bf16.h>

typedef __attribute__((ext_vector_type(16))) _Float16 v16h;
typedef __attribute__((ext_vector_type(8)))  _Float16 v8h;
typedef __attribute__((ext_vector_type(8)))  float    v8f;

#define MASS_F   2.8f
#define DT_F     0.016f
#define RADIUS_F 0.09f
#define S_STEPS  1024
#define B_BODIES 8192

__device__ __forceinline__ float softplusf(float x) {
    return (x > 20.f) ? x : log1pf(expf(x));
}
__device__ __forceinline__ float sgnf(float v) {
    return (v > 0.f) ? 1.f : ((v < 0.f) ? -1.f : 0.f);
}

// gfx1250 has V_TANH_F32 as a native transcendental.
__device__ __forceinline__ float fast_tanhf(float x) {
#if __has_builtin(__builtin_amdgcn_tanhf)
    return __builtin_amdgcn_tanhf(x);
#else
    float y;
    asm("v_tanh_f32 %0, %1" : "=v"(y) : "v"(x));
    return y;
#endif
}

// pack two f32 -> packed f16 dword
__device__ __forceinline__ unsigned packh(float a, float b) {
    _Float16 ha = (_Float16)a, hb = (_Float16)b;
    unsigned short ua = __builtin_bit_cast(unsigned short, ha);
    unsigned short ub = __builtin_bit_cast(unsigned short, hb);
    return (unsigned)ua | ((unsigned)ub << 16);
}

union HV16 { unsigned u[8]; v16h v; };

__global__ __launch_bounds__(32) void omni_rollout_kernel(
    const float* __restrict__ init_state,   // (8192, 6)
    const float* __restrict__ cmd_seq,      // (8192, 1024, 3)
    const float* __restrict__ W1, const float* __restrict__ b1,   // (64,6),(64)
    const float* __restrict__ W2, const float* __restrict__ b2,   // (64,64),(64)
    const float* __restrict__ W3, const float* __restrict__ b3,   // (3,64),(3)
    const float* __restrict__ com_offset,
    const float* __restrict__ inertia_raw,
    const float* __restrict__ gain_raw,
    const float* __restrict__ grip_raw,
    const float* __restrict__ dragv_raw,
    const float* __restrict__ dragc_raw,
    float* __restrict__ out)                // (8192, 1025, 6)
{
    __shared__ __align__(16) _Float16 s_h [32 * 64];   // hidden activations
    __shared__ __align__(16) float    s_res[32 * 4];   // MLP residual output

    const int lane = threadIdx.x;       // 0..31 (wave32)
    const int n    = lane & 15;         // WMMA column / M-row within tile
    const int hi   = lane >> 4;         // lane half
    const int b    = blockIdx.x * 32 + lane;

    // ---------------- physics parameters (once) ----------------
    const float inertia = softplusf(inertia_raw[0]) + 1e-4f;
    const float gain    = softplusf(gain_raw[0]);
    const float grip    = softplusf(grip_raw[0]);
    const float dv0 = softplusf(dragv_raw[0]), dv1 = softplusf(dragv_raw[1]), dv2 = softplusf(dragv_raw[2]);
    const float dc0 = softplusf(dragc_raw[0]), dc1 = softplusf(dragc_raw[1]), dc2 = softplusf(dragc_raw[2]);
    const float dx = com_offset[0], dy = com_offset[1];
    const float inv_inertia = 1.0f / inertia;
    const float inv_mass    = 1.0f / MASS_F;

    // wheel chain collapses: F_motor = gain * M * (cmd - vel), M = A^T A, A rows (fx,fy,0.09)
    float M00 = 0.f, M01 = 0.f, M02 = 0.f, M11 = 0.f, M12 = 0.f, M22 = 0.f;
    {
        const float degs[4] = {60.f, 130.f, 230.f, 300.f};
        #pragma unroll
        for (int w = 0; w < 4; w++) {
            float a  = degs[w] * (3.14159265358979323846f / 180.f);
            float fx = -sinf(a), fy = cosf(a), ar = RADIUS_F;
            M00 += fx * fx; M01 += fx * fy; M02 += fx * ar;
            M11 += fy * fy; M12 += fy * ar; M22 += ar * ar;
        }
    }

    // ---------------- resident WMMA B fragments (weights) ----------------
    // B (32x16) layout: lane n = output column, K = hi*16 + e (K pairs per VGPR)
    // Layer-1 bias folded into K-slot 6: B1[k=6][n] = b1[n], A1[m][6] = 1.0
    const int kb = hi * 16;
    v16h B1[4], B2[2][4], B3[2];
    #pragma unroll
    for (int t = 0; t < 4; t++) {
        v16h f;
        #pragma unroll
        for (int e = 0; e < 16; e++) {
            int k = kb + e;
            _Float16 w = (_Float16)0.0f;
            if (k < 6)       w = (_Float16)W1[(t * 16 + n) * 6 + k];
            else if (k == 6) w = (_Float16)b1[t * 16 + n];     // bias row
            f[e] = w;
        }
        B1[t] = f;
    }
    #pragma unroll
    for (int c = 0; c < 2; c++)
        #pragma unroll
        for (int t = 0; t < 4; t++) {
            v16h f;
            #pragma unroll
            for (int e = 0; e < 16; e++)
                f[e] = (_Float16)W2[(t * 16 + n) * 64 + (c * 32 + kb + e)];
            B2[c][t] = f;
        }
    #pragma unroll
    for (int c = 0; c < 2; c++) {
        v16h f;
        #pragma unroll
        for (int e = 0; e < 16; e++)
            f[e] = (n < 3) ? (_Float16)W3[n * 64 + c * 32 + kb + e] : (_Float16)0.0f;
        B3[c] = f;
    }
    float b2c[4];
    #pragma unroll
    for (int t = 0; t < 4; t++) b2c[t] = b2[t * 16 + n];
    const float b3c = (n < 3) ? b3[n] : 0.0f;

    // ---------------- state ----------------
    float px = init_state[b * 6 + 0];
    float py = init_state[b * 6 + 1];
    float th = init_state[b * 6 + 2];
    float vx = init_state[b * 6 + 3];
    float vy = init_state[b * 6 + 4];
    float om = init_state[b * 6 + 5];

    {   // trajectory row 0 = initial state
        float2* o = reinterpret_cast<float2*>(out + (size_t)b * 1025u * 6u);
        o[0] = make_float2(px, py); o[1] = make_float2(th, vx); o[2] = make_float2(vy, om);
    }

    const float* cmdp = cmd_seq + (size_t)b * S_STEPS * 3;
    const unsigned ONE_H = 0x00003C00u;   // (f16)1.0 in low half: bias K-slot 6

    #pragma unroll 1
    for (int s = 0; s < S_STEPS; s++) {
        const float cx = cmdp[s * 3 + 0];
        const float cy = cmdp[s * 3 + 1];
        const float cz = cmdp[s * 3 + 2];
        if (s + 1 < S_STEPS) __builtin_prefetch(cmdp + (s + 1) * 3, 0, 0);

        // ---- layer-1 A fragments, no LDS: lane m of tile0 == body m (own regs);
        //      tile1 bodies live in partner lane-half -> 3x shfl_xor(16).
        const unsigned d0 = packh(vx, vy);
        const unsigned d1 = packh(om, cx);
        const unsigned d2 = packh(cy, cz);
        const unsigned p0 = __shfl_xor(d0, 16, 32);
        const unsigned p1 = __shfl_xor(d1, 16, 32);
        const unsigned p2 = __shfl_xor(d2, 16, 32);

        v16h A1[2];
        {
            HV16 a0, a1;
            #pragma unroll
            for (int i = 0; i < 8; i++) { a0.u[i] = 0u; a1.u[i] = 0u; }
            const bool lo = (hi == 0);        // A layout: lanes 0-15 hold K=0..7
            a0.u[0] = lo ? d0 : 0u;  a0.u[1] = lo ? d1 : 0u;
            a0.u[2] = lo ? d2 : 0u;  a0.u[3] = lo ? ONE_H : 0u;
            a1.u[0] = lo ? p0 : 0u;  a1.u[1] = lo ? p1 : 0u;
            a1.u[2] = lo ? p2 : 0u;  a1.u[3] = lo ? ONE_H : 0u;
            A1[0] = a0.v; A1[1] = a1.v;
        }

        #pragma unroll
        for (int mt = 0; mt < 2; mt++) {
            #pragma unroll
            for (int t = 0; t < 4; t++) {
                v8f acc;
                #pragma unroll
                for (int r = 0; r < 8; r++) acc[r] = 0.0f;
                acc = __builtin_amdgcn_wmma_f32_16x16x32_f16(
                        false, A1[mt], false, B1[t], (short)0, acc, false, false);
                const int rowb = mt * 16 + hi * 8;
                #pragma unroll
                for (int r = 0; r < 8; r++)
                    s_h[(rowb + r) * 64 + t * 16 + n] = (_Float16)fast_tanhf(acc[r]);
            }
        }
        __syncthreads();

        // ---- layer 2: A = 16x64 (two K-chunks), ds_read_b128 per chunk-half ----
        v16h A2[2][2];
        #pragma unroll
        for (int mt = 0; mt < 2; mt++) {
            const _Float16* rp = &s_h[(mt * 16 + n) * 64];
            #pragma unroll
            for (int c = 0; c < 2; c++) {
                v8h lo8 = *reinterpret_cast<const v8h*>(rp + c * 32 + hi * 16);
                v8h hh8 = *reinterpret_cast<const v8h*>(rp + c * 32 + hi * 16 + 8);
                v16h a;
                #pragma unroll
                for (int e = 0; e < 8; e++) { a[e] = lo8[e]; a[e + 8] = hh8[e]; }
                A2[mt][c] = a;
            }
        }
        __syncthreads();
        #pragma unroll
        for (int mt = 0; mt < 2; mt++) {
            #pragma unroll
            for (int t = 0; t < 4; t++) {
                v8f acc;
                #pragma unroll
                for (int r = 0; r < 8; r++) acc[r] = 0.0f;
                acc = __builtin_amdgcn_wmma_f32_16x16x32_f16(
                        false, A2[mt][0], false, B2[0][t], (short)0, acc, false, false);
                acc = __builtin_amdgcn_wmma_f32_16x16x32_f16(
                        false, A2[mt][1], false, B2[1][t], (short)0, acc, false, false);
                const int rowb = mt * 16 + hi * 8;
                #pragma unroll
                for (int r = 0; r < 8; r++)
                    s_h[(rowb + r) * 64 + t * 16 + n] = (_Float16)fast_tanhf(acc[r] + b2c[t]);
            }
        }
        __syncthreads();

        // ---- layer 3: 64 -> 3 (single N tile, columns 0..2 valid) ----
        v16h A3[2][2];
        #pragma unroll
        for (int mt = 0; mt < 2; mt++) {
            const _Float16* rp = &s_h[(mt * 16 + n) * 64];
            #pragma unroll
            for (int c = 0; c < 2; c++) {
                v8h lo8 = *reinterpret_cast<const v8h*>(rp + c * 32 + hi * 16);
                v8h hh8 = *reinterpret_cast<const v8h*>(rp + c * 32 + hi * 16 + 8);
                v16h a;
                #pragma unroll
                for (int e = 0; e < 8; e++) { a[e] = lo8[e]; a[e + 8] = hh8[e]; }
                A3[mt][c] = a;
            }
        }
        #pragma unroll
        for (int mt = 0; mt < 2; mt++) {
            v8f acc;
            #pragma unroll
            for (int r = 0; r < 8; r++) acc[r] = 0.0f;
            acc = __builtin_amdgcn_wmma_f32_16x16x32_f16(
                    false, A3[mt][0], false, B3[0], (short)0, acc, false, false);
            acc = __builtin_amdgcn_wmma_f32_16x16x32_f16(
                    false, A3[mt][1], false, B3[1], (short)0, acc, false, false);
            if (n < 3) {
                const int rowb = mt * 16 + hi * 8;
                #pragma unroll
                for (int r = 0; r < 8; r++)
                    s_res[(rowb + r) * 4 + n] = acc[r] + b3c;
            }
        }
        __syncthreads();
        const float r0 = s_res[lane * 4 + 0];
        const float r1 = s_res[lane * 4 + 1];
        const float r2 = s_res[lane * 4 + 2];
        __syncthreads();

        // ---- physics (fp32 VALU) ----
        const float ex = cx - vx, ey = cy - vy, ez = cz - om;
        float Fx  = gain * (M00 * ex + M01 * ey + M02 * ez) + grip * ex - dv0 * vx - dc0 * sgnf(vx);
        float Fy  = gain * (M01 * ex + M11 * ey + M12 * ez) + grip * ey - dv1 * vy - dc1 * sgnf(vy);
        float Tau = gain * (M02 * ex + M12 * ey + M22 * ez) + grip * ez - dv2 * om - dc2 * sgnf(om);

        const float tau_com   = Tau - (dx * Fy - dy * Fx);
        const float acc_alpha = tau_com * inv_inertia;
        const float osq       = om * om;
        const float acc_x = Fx * inv_mass - acc_alpha * dy - osq * dx;
        const float acc_y = Fy * inv_mass + acc_alpha * dx - osq * dy;

        const float nvx = vx + (acc_x + r0) * DT_F;
        const float nvy = vy + (acc_y + r1) * DT_F;
        const float nom = om + (acc_alpha + r2) * DT_F;

        // theta update: native sin/cos (|angle| <~ pi + eps, well inside native range),
        // branchless wrap to (-pi, pi] instead of atan2f(sin, cos).
        const float ntp = th + nom * DT_F;
        const float s0 = __sinf(ntp), c0 = __cosf(ntp);
        const float nth = ntp - 6.28318530717958647692f *
                                rintf(ntp * 0.15915494309189533577f);
        const float npx = px + (c0 * nvx - s0 * nvy) * DT_F;
        const float npy = py + (s0 * nvx + c0 * nvy) * DT_F;

        px = npx; py = npy; th = nth; vx = nvx; vy = nvy; om = nom;

        float2* o = reinterpret_cast<float2*>(out + ((size_t)b * 1025u + (size_t)(s + 1)) * 6u);
        o[0] = make_float2(px, py); o[1] = make_float2(th, vx); o[2] = make_float2(vy, om);
    }
}

extern "C" void kernel_launch(void* const* d_in, const int* in_sizes, int n_in,
                              void* d_out, int out_size, void* d_ws, size_t ws_size,
                              hipStream_t stream) {
    const float* init_state = (const float*)d_in[0];
    const float* cmd_seq    = (const float*)d_in[1];
    const float* W1 = (const float*)d_in[2];
    const float* b1 = (const float*)d_in[3];
    const float* W2 = (const float*)d_in[4];
    const float* b2 = (const float*)d_in[5];
    const float* W3 = (const float*)d_in[6];
    const float* b3 = (const float*)d_in[7];
    const float* com_offset  = (const float*)d_in[8];
    const float* inertia_raw = (const float*)d_in[9];
    const float* gain_raw    = (const float*)d_in[10];
    const float* grip_raw    = (const float*)d_in[11];
    const float* dragv_raw   = (const float*)d_in[12];
    const float* dragc_raw   = (const float*)d_in[13];
    float* out = (float*)d_out;

    dim3 grid(B_BODIES / 32);
    dim3 block(32);
    hipLaunchKernelGGL(omni_rollout_kernel, grid, block, 0, stream,
                       init_state, cmd_seq, W1, b1, W2, b2, W3, b3,
                       com_offset, inertia_raw, gain_raw, grip_raw,
                       dragv_raw, dragc_raw, out);
}
